// PointPillars_15058155340475
// MI455X (gfx1250) — matrix operand order repaired and checked by reading, not compiled, not run on hardware
//
#include <hip/hip_runtime.h>
#include <hip/hip_bf16.h>
#include <stdint.h>

// ---------------------------------------------------------------------------
// PointPillars forward for gfx1250 (CDNA5, wave32, WMMA).
//   PFN (VALU) -> scatter f16 BEV (NHWC)
//   -> conv1/conv2: LDS-staged input tiles + v_wmma_f32_16x16x32_f16
//      (block = 8 waves x 16 out-ch, 64 output pixels; B fragment reused 4x)
//   -> 1x1 heads via WMMA, NCHW f32 output.
// ---------------------------------------------------------------------------

typedef __attribute__((ext_vector_type(16))) _Float16 v16h;
typedef __attribute__((ext_vector_type(8)))  _Float16 v8h;
typedef __attribute__((ext_vector_type(8)))  float    v8f;

#define N_PILLARS 20000
#define P_PTS     32
#define F_IN      7
#define C_EMB     64
#define HW_IN     512
#define HW_OUT    256
#define C1        128
#define N_HEAD    34        // 20 cls + 14 box
#define N_HEAD_PAD 48       // padded to 3 WMMA N-tiles

__device__ __forceinline__ v16h pack16(v8h lo, v8h hi) {
    v16h r;
#pragma unroll
    for (int i = 0; i < 8; ++i) { r[i] = lo[i]; r[8 + i] = hi[i]; }
    return r;
}

// ------------------------- workspace zeroing -------------------------------
__global__ void zero_u4_kernel(uint4* __restrict__ p, int n) {
    int i = blockIdx.x * blockDim.x + threadIdx.x;
    if (i < n) p[i] = make_uint4(0u, 0u, 0u, 0u);
}

// ------------------------- weight repacking --------------------------------
// OIHW f32 -> [o][(ky*3+kx)*C + c] f16   (B-matrix friendly: K contiguous per row)
__global__ void pack_conv_w_kernel(const float* __restrict__ w, _Float16* __restrict__ wp,
                                   int O, int C) {
    int i = blockIdx.x * blockDim.x + threadIdx.x;
    int total = O * C * 9;
    if (i >= total) return;
    int o   = i / (C * 9);
    int rem = i % (C * 9);
    int tap = rem / C;           // ky*3 + kx
    int c   = rem % C;
    int ky = tap / 3, kx = tap % 3;
    float v = w[(((size_t)o * C + c) * 3 + ky) * 3 + kx];
    wp[(size_t)o * (9 * C) + (size_t)tap * C + c] = (_Float16)v;
}

// Heads: w_cls (20x128) + w_box (14x128) -> [48][128] f16 (rows 34..47 zero), bias f32[48]
__global__ void pack_head_kernel(const float* __restrict__ wcls, const float* __restrict__ bcls,
                                 const float* __restrict__ wbox, const float* __restrict__ bbox,
                                 _Float16* __restrict__ wh, float* __restrict__ bh) {
    int i = blockIdx.x * blockDim.x + threadIdx.x;
    if (i >= N_HEAD_PAD * C1) return;
    int n = i / C1, c = i % C1;
    float v = 0.0f;
    if (n < 20)      v = wcls[n * C1 + c];
    else if (n < 34) v = wbox[(n - 20) * C1 + c];
    wh[i] = (_Float16)v;
    if (c == 0) {
        float b = 0.0f;
        if (n < 20)      b = bcls[n];
        else if (n < 34) b = bbox[n - 20];
        bh[n] = b;
    }
}

// ------------------------- PFN + masked max + scatter ----------------------
// One block per pillar, thread c in [0,64) = embedding channel.
__global__ void pfn_scatter_kernel(const float* __restrict__ feat,          // [N][P][F]
                                   const unsigned char* __restrict__ mask,  // [N][P]
                                   const int* __restrict__ coords,          // [N][2] (x,y)
                                   const float* __restrict__ w_pfn,         // [F][64]
                                   const float* __restrict__ b_pfn,         // [64]
                                   _Float16* __restrict__ bev)              // [512][512][64]
{
    const int n = blockIdx.x;
    const int c = threadIdx.x;            // 0..63
    if (n >= N_PILLARS) return;

    float wc[F_IN];
#pragma unroll
    for (int f = 0; f < F_IN; ++f) wc[f] = w_pfn[f * C_EMB + c];
    const float bc = b_pfn[c];

    float best = -1.0e9f;
    bool  any  = false;
    const float* fb = feat + (size_t)n * P_PTS * F_IN;
    const unsigned char* mb = mask + (size_t)n * P_PTS;
#pragma unroll 4
    for (int p = 0; p < P_PTS; ++p) {
        if (mb[p]) {
            any = true;
            float s = bc;
#pragma unroll
            for (int f = 0; f < F_IN; ++f) s += fb[p * F_IN + f] * wc[f];
            s = s > 0.0f ? s : 0.0f;      // relu before masked max
            best = s > best ? s : best;
        }
    }
    const float emb = any ? best : 0.0f;

    const int x = coords[n * 2 + 0];
    const int y = coords[n * 2 + 1];
    if (x >= 0 && x < HW_IN && y >= 0 && y < HW_IN)
        bev[((size_t)y * HW_IN + x) * C_EMB + c] = (_Float16)emb;
}

// ------------------------- 3x3 conv via LDS + WMMA -------------------------
// Block = 256 threads (8 waves). Block computes 64 output pixels (one row strip)
// x all 128 output channels. Wave nt owns N-tile of 16 channels and keeps 4
// accumulators (4 M-tiles of 16 pixels); one B fragment feeds 4 WMMAs.
// Input tile (3 rows x (64*STRIDE+2) cols x CIN, zero-padded) staged in LDS.
template <int CIN, int IN_HW, int STRIDE, int PADB>
__global__ __launch_bounds__(256)
void conv3x3_wmma_kernel(const _Float16* __restrict__ in,      // [IN_HW][IN_HW][CIN]
                         const _Float16* __restrict__ wpack,   // [128][9*CIN]
                         const float* __restrict__ bias,       // [128]
                         _Float16* __restrict__ out)           // [256][256][128]
{
    constexpr int MPIX   = 64;
    constexpr int TILE_W = MPIX * STRIDE + 2;
    constexpr int ROW_E  = TILE_W * CIN;         // f16 elements per tile row
    __shared__ __align__(16) _Float16 tile[3 * ROW_E];

    const int tid  = threadIdx.x;
    const int lane = tid & 31;
    const int nt   = tid >> 5;                   // wave id = N-tile (0..7)
    const int ox0  = blockIdx.x * MPIX;
    const int oy   = blockIdx.y;
    const int iy0  = oy * STRIDE - PADB;
    const int ix0  = ox0 * STRIDE - PADB;

    // ---- stage input tile into LDS (zero-padded at image borders) ----
    constexpr int COUNT8 = 3 * ROW_E / 8;
    for (int i = tid; i < COUNT8; i += 256) {
        const int idx8 = i * 8;
        const int r    = idx8 / ROW_E;
        const int rem  = idx8 - r * ROW_E;
        const int col  = rem / CIN;
        const int c    = rem - col * CIN;        // multiple of 8
        const int iy = iy0 + r, ix = ix0 + col;
        v8h v = {};
        if (iy >= 0 && iy < IN_HW && ix >= 0 && ix < IN_HW)
            v = *(const v8h*)(in + ((size_t)iy * IN_HW + ix) * CIN + c);
        *(v8h*)&tile[idx8] = v;
    }
    __syncthreads();

    // ---- WMMA main loop ----
    const int  mrow = lane & 15;                 // M row / N column of this lane
    const bool hi   = lane >= 16;
    const int  K    = 9 * CIN;

    v8f acc[4] = {};
    const _Float16* wrow = wpack + (size_t)(nt * 16 + mrow) * K + (hi ? 16 : 0);

#pragma unroll
    for (int ky = 0; ky < 3; ++ky) {
#pragma unroll
        for (int kx = 0; kx < 3; ++kx) {
            const int tap = ky * 3 + kx;
#pragma unroll
            for (int cb = 0; cb < CIN; cb += 32) {
                // B fragment (32x16): lane<16 -> K 0..15; lane>=16 -> K 16..31 (contiguous)
                const _Float16* bp = wrow + tap * CIN + cb;
                const v16h b = pack16(*(const v8h*)(bp), *(const v8h*)(bp + 8));
#pragma unroll
                for (int s = 0; s < 4; ++s) {
                    // A fragment (16x32) from LDS: lane<16 -> K 0..15; hi -> K 16..31 (contiguous)
                    const int col = (s * 16 + mrow) * STRIDE + kx;
                    const _Float16* ap = &tile[(ky * TILE_W + col) * CIN + cb + (hi ? 16 : 0)];
                    const v16h a = pack16(*(const v8h*)(ap), *(const v8h*)(ap + 8));
                    acc[s] = __builtin_amdgcn_wmma_f32_16x16x32_f16(
                                 false, a, false, b, (short)0, acc[s], false, false);
                }
            }
        }
    }

    // ---- bias + relu + NHWC f16 store. acc[s] reg r: M = s*16 + r + (hi?8:0) ----
    const int oc = nt * 16 + mrow;
    const float bv = bias[oc];
#pragma unroll
    for (int s = 0; s < 4; ++s) {
#pragma unroll
        for (int r = 0; r < 8; ++r) {
            const int m = s * 16 + r + (hi ? 8 : 0);
            float v = acc[s][r] + bv;
            v = v > 0.0f ? v : 0.0f;
            out[((size_t)oy * HW_OUT + ox0 + m) * C1 + oc] = (_Float16)v;
        }
    }
}

// ------------------------- 1x1 heads via WMMA ------------------------------
// M = 16 consecutive pixels (flat 256*256), N = 16 head channels (3 tiles, 34 valid), K=128.
__global__ void head_wmma_kernel(const _Float16* __restrict__ feat,   // [65536][128]
                                 const _Float16* __restrict__ whead,  // [48][128]
                                 const float* __restrict__ bhead,     // [48]
                                 float* __restrict__ out)             // [34][65536] NCHW
{
    const int lane = threadIdx.x;
    const int pix0 = blockIdx.x * 16;
    const int nt   = blockIdx.y;             // 0..2
    const int mrow = lane & 15;
    const bool hi  = lane >= 16;

    const _Float16* abase = feat + (size_t)(pix0 + mrow) * C1;
    v8f acc = {};
#pragma unroll
    for (int cb = 0; cb < C1; cb += 32) {
        const _Float16* ap = abase + cb + (hi ? 16 : 0);
        const v16h a = pack16(*(const v8h*)(ap), *(const v8h*)(ap + 8));
        const _Float16* bp = whead + (size_t)(nt * 16 + mrow) * C1 + cb + (hi ? 16 : 0);
        const v16h b = pack16(*(const v8h*)(bp), *(const v8h*)(bp + 8));
        acc = __builtin_amdgcn_wmma_f32_16x16x32_f16(false, a, false, b,
                                                     (short)0, acc, false, false);
    }

    const int oc = nt * 16 + mrow;
    if (oc < N_HEAD) {
        const float bv = bhead[oc];
#pragma unroll
        for (int r = 0; r < 8; ++r) {
            const int m = r + (hi ? 8 : 0);
            out[(size_t)oc * (HW_OUT * HW_OUT) + pix0 + m] = acc[r] + bv;
        }
    }
}

// ---------------------------------------------------------------------------
extern "C" void kernel_launch(void* const* d_in, const int* in_sizes, int n_in,
                              void* d_out, int out_size, void* d_ws, size_t ws_size,
                              hipStream_t stream) {
    (void)in_sizes; (void)n_in; (void)out_size; (void)ws_size;

    const float*         feat   = (const float*)d_in[0];
    const unsigned char* mask   = (const unsigned char*)d_in[1];
    const int*           coords = (const int*)d_in[2];
    // d_in[3]=H, d_in[4]=W (compile-time 512 here)
    const float* w_pfn = (const float*)d_in[5];
    const float* b_pfn = (const float*)d_in[6];
    const float* w1    = (const float*)d_in[7];
    const float* b1    = (const float*)d_in[8];
    const float* w2    = (const float*)d_in[9];
    const float* b2    = (const float*)d_in[10];
    const float* w_cls = (const float*)d_in[11];
    const float* b_cls = (const float*)d_in[12];
    const float* w_box = (const float*)d_in[13];
    const float* b_box = (const float*)d_in[14];
    float* out = (float*)d_out;

    // ---- workspace layout (bytes) ----
    char* ws = (char*)d_ws;
    const size_t BEV_BYTES   = (size_t)HW_IN * HW_IN * C_EMB * 2;     // 32 MB
    const size_t FEAT1_BYTES = (size_t)HW_OUT * HW_OUT * C1 * 2;      // 16 MB
    const size_t W1P_BYTES   = (size_t)C1 * (9 * C_EMB) * 2;          // 144 KB
    const size_t W2P_BYTES   = (size_t)C1 * (9 * C1) * 2;             // 288 KB
    const size_t WH_BYTES    = (size_t)N_HEAD_PAD * C1 * 2;

    _Float16* bev    = (_Float16*)(ws);
    _Float16* feat1  = (_Float16*)(ws + BEV_BYTES);
    _Float16* w1pack = (_Float16*)(ws + BEV_BYTES + FEAT1_BYTES);
    _Float16* w2pack = (_Float16*)(ws + BEV_BYTES + FEAT1_BYTES + W1P_BYTES);
    _Float16* whead  = (_Float16*)(ws + BEV_BYTES + FEAT1_BYTES + W1P_BYTES + W2P_BYTES);
    float*    bhead  = (float*)   (ws + BEV_BYTES + FEAT1_BYTES + W1P_BYTES + W2P_BYTES + WH_BYTES);
    _Float16* feat2  = bev;   // reuse: conv1 has fully consumed the BEV before conv2 writes

    // 1) zero BEV (f16)
    {
        int n16 = (int)(BEV_BYTES / 16);
        zero_u4_kernel<<<(n16 + 255) / 256, 256, 0, stream>>>((uint4*)bev, n16);
    }
    // 2) repack weights to f16 GEMM layout
    {
        int t1 = C1 * C_EMB * 9;
        pack_conv_w_kernel<<<(t1 + 255) / 256, 256, 0, stream>>>(w1, w1pack, C1, C_EMB);
        int t2 = C1 * C1 * 9;
        pack_conv_w_kernel<<<(t2 + 255) / 256, 256, 0, stream>>>(w2, w2pack, C1, C1);
        int th = N_HEAD_PAD * C1;
        pack_head_kernel<<<(th + 255) / 256, 256, 0, stream>>>(w_cls, b_cls, w_box, b_box,
                                                               whead, bhead);
    }
    // 3) PFN + masked max + scatter into BEV
    pfn_scatter_kernel<<<N_PILLARS, C_EMB, 0, stream>>>(feat, mask, coords, w_pfn, b_pfn, bev);

    // 4) conv1: 64 -> 128, 3x3, stride 2, SAME (pad_beg 0) ; 512 -> 256, relu
    {
        dim3 grid(HW_OUT / 64, HW_OUT);
        conv3x3_wmma_kernel<C_EMB, HW_IN, 2, 0><<<grid, 256, 0, stream>>>(bev, w1pack, b1, feat1);
    }
    // 5) conv2: 128 -> 128, 3x3, stride 1, SAME (pad 1) ; 256 -> 256, relu
    {
        dim3 grid(HW_OUT / 64, HW_OUT);
        conv3x3_wmma_kernel<C1, HW_OUT, 1, 1><<<grid, 256, 0, stream>>>(feat1, w2pack, b2, feat2);
    }
    // 6) 1x1 heads (cls|box concat), NCHW f32 output
    {
        dim3 grid((HW_OUT * HW_OUT) / 16, N_HEAD_PAD / 16);
        head_wmma_kernel<<<grid, 32, 0, stream>>>(feat2, whead, bhead, out);
    }
}
